// GCN_32573031973225
// MI455X (gfx1250) — matrix operand order repaired and checked by reading, not compiled, not run on hardware
//
#include <hip/hip_runtime.h>
#include <math.h>

typedef __attribute__((ext_vector_type(16))) _Float16 v16h;
typedef __attribute__((ext_vector_type(8)))  float    v8f;

#define IN_DIM  512
#define HID_DIM 128
#define OUT_DIM 64

// ---------------------------------------------------------------------------
// Degree / normalization
// ---------------------------------------------------------------------------
__global__ void deg_init_kernel(float* __restrict__ deg, int n) {
    int i = blockIdx.x * blockDim.x + threadIdx.x;
    if (i < n) deg[i] = 1.0f;  // self-loop contributes 1
}

__global__ void deg_accum_kernel(const int* __restrict__ dst, float* __restrict__ deg, int e) {
    int i = blockIdx.x * blockDim.x + threadIdx.x;
    if (i < e) atomicAdd(&deg[dst[i]], 1.0f);
}

__global__ void deg_rsqrt_kernel(float* __restrict__ deg, int n) {
    int i = blockIdx.x * blockDim.x + threadIdx.x;
    if (i < n) deg[i] = rsqrtf(deg[i]);  // deg -> dis
}

// ---------------------------------------------------------------------------
// WMMA GEMM: H[M,N] = X[M,K] @ W[K,N], fp32 in/out, f16 multiply, f32 accum.
// One wave computes one 16x16 output tile; K consumed 32 at a time.
// A (16x32 f16) layout: lane l (m = l&15, g = l>>4), half j in 0..7 -> K = g*8+j,
//                       half 8+j -> K = 16+g*8+j.
// B (32x16 f16) layout: lane l (n = l&15, g = l>>4), half j in 0..7 -> K = g*8+j,
//                       half 8+j -> K = 16+g*8+j.
// C/D (16x16 f32) layout: vgpr r, lane l -> row = r + g*8, col = l&15.
// ---------------------------------------------------------------------------
__global__ __launch_bounds__(256) void gemm_wmma_f16(
    const float* __restrict__ X, const float* __restrict__ W,
    float* __restrict__ H, int M, int K, int N)
{
    int wave = (int)((blockIdx.x * blockDim.x + threadIdx.x) >> 5);
    int lane = threadIdx.x & 31;
    int ntiles = N >> 4;
    int m0 = (wave / ntiles) << 4;
    int n0 = (wave % ntiles) << 4;
    if (m0 >= M) return;

    int l16 = lane & 15;
    int g   = lane >> 4;
    const float* xrow = X + (size_t)(m0 + l16) * (size_t)K;

    v8f acc = {};
    for (int k0 = 0; k0 < K; k0 += 32) {
        v16h a, b;
#pragma unroll
        for (int j = 0; j < 8; ++j) {
            a[j]     = (_Float16)xrow[k0 + g * 8 + j];
            a[8 + j] = (_Float16)xrow[k0 + 16 + g * 8 + j];
        }
#pragma unroll
        for (int h = 0; h < 16; ++h) {
            b[h] = (_Float16)W[(size_t)(k0 + (h >> 3) * 16 + g * 8 + (h & 7)) * (size_t)N + (n0 + l16)];
        }
        acc = __builtin_amdgcn_wmma_f32_16x16x32_f16(
            /*neg_a=*/false, a, /*neg_b=*/false, b,
            /*c_mod=*/(short)0, acc, /*reuse_a=*/false, /*reuse_b=*/false);
    }
#pragma unroll
    for (int r = 0; r < 8; ++r) {
        H[(size_t)(m0 + r + g * 8) * (size_t)N + (n0 + l16)] = acc[r];
    }
}

// ---------------------------------------------------------------------------
// Edge scatter, F=128: one wave per edge, float4 per lane, fp32 atomics to L2.
// ---------------------------------------------------------------------------
__global__ __launch_bounds__(256) void scatter_f128(
    const int* __restrict__ src, const int* __restrict__ dst,
    const float* __restrict__ dis, const float* __restrict__ H,
    float* __restrict__ G, int E)
{
    int e = (int)((blockIdx.x * blockDim.x + threadIdx.x) >> 5);
    if (e >= E) return;
    int lane = threadIdx.x & 31;
    int s = src[e], d = dst[e];
    float w = dis[s] * dis[d];
    float4 v = ((const float4*)(H + (size_t)s * 128))[lane];
    float* g = G + (size_t)d * 128 + lane * 4;
    atomicAdd(g + 0, v.x * w);
    atomicAdd(g + 1, v.y * w);
    atomicAdd(g + 2, v.z * w);
    atomicAdd(g + 3, v.w * w);
}

// Edge scatter, F=64: 16 lanes per edge (one float4 chunk per lane).
__global__ __launch_bounds__(256) void scatter_f64(
    const int* __restrict__ src, const int* __restrict__ dst,
    const float* __restrict__ dis, const float* __restrict__ H,
    float* __restrict__ G, int E)
{
    int t = blockIdx.x * blockDim.x + threadIdx.x;
    int e = t >> 4;
    if (e >= E) return;
    int c = t & 15;
    int s = src[e], d = dst[e];
    float w = dis[s] * dis[d];
    float4 v = ((const float4*)(H + (size_t)s * 64))[c];
    float* g = G + (size_t)d * 64 + c * 4;
    atomicAdd(g + 0, v.x * w);
    atomicAdd(g + 1, v.y * w);
    atomicAdd(g + 2, v.z * w);
    atomicAdd(g + 3, v.w * w);
}

// ---------------------------------------------------------------------------
// Post layer 1: g1 = relu(g1 + h1 * dis^2 (self-loop) + b1)
// ---------------------------------------------------------------------------
__global__ void post1_kernel(float* __restrict__ G, const float* __restrict__ H,
                             const float* __restrict__ dis, const float* __restrict__ b1,
                             int total)
{
    int idx = blockIdx.x * blockDim.x + threadIdx.x;
    if (idx >= total) return;
    int i = idx >> 7;          // / 128
    int f = idx & 127;
    float di = dis[i];
    float v = G[idx] + H[idx] * di * di + b1[f];
    G[idx] = v > 0.0f ? v : 0.0f;
}

// Initialize output with self-loop term + bias (also re-inits poisoned d_out).
__global__ void init_out_kernel(float* __restrict__ O, const float* __restrict__ H2,
                                const float* __restrict__ dis, const float* __restrict__ b2,
                                int total)
{
    int idx = blockIdx.x * blockDim.x + threadIdx.x;
    if (idx >= total) return;
    int i = idx >> 6;          // / 64
    int f = idx & 63;
    float di = dis[i];
    O[idx] = H2[idx] * di * di + b2[f];
}

// ---------------------------------------------------------------------------
// In-place log_softmax over 64 columns: one wave per row (wave32).
// ---------------------------------------------------------------------------
__global__ __launch_bounds__(256) void logsoftmax64_kernel(float* __restrict__ O, int M)
{
    int row = (int)((blockIdx.x * blockDim.x + threadIdx.x) >> 5);
    if (row >= M) return;
    int lane = threadIdx.x & 31;
    float* p = O + (size_t)row * 64;
    float v0 = p[lane];
    float v1 = p[lane + 32];
    float m = fmaxf(v0, v1);
#pragma unroll
    for (int off = 16; off > 0; off >>= 1)
        m = fmaxf(m, __shfl_xor(m, off, 32));
    float s = expf(v0 - m) + expf(v1 - m);
#pragma unroll
    for (int off = 16; off > 0; off >>= 1)
        s += __shfl_xor(s, off, 32);
    float lse = m + logf(s);
    p[lane]      = v0 - lse;
    p[lane + 32] = v1 - lse;
}

// ---------------------------------------------------------------------------
extern "C" void kernel_launch(void* const* d_in, const int* in_sizes, int n_in,
                              void* d_out, int out_size, void* d_ws, size_t ws_size,
                              hipStream_t stream)
{
    const float* x  = (const float*)d_in[0];
    const int*   ei = (const int*)d_in[1];
    const float* W1 = (const float*)d_in[2];
    const float* b1 = (const float*)d_in[3];
    const float* W2 = (const float*)d_in[4];
    const float* b2 = (const float*)d_in[5];
    float* out = (float*)d_out;

    const int E = in_sizes[1] / 2;
    const int N = in_sizes[0] / IN_DIM;   // 100000 (divisible by 16)
    const int* src = ei;
    const int* dst = ei + E;

    // Workspace carve-out (~128.4 MB total)
    char* ws = (char*)d_ws;
    size_t off = 0;
    auto carve = [&](size_t bytes) -> void* {
        void* p = ws + off;
        off += bytes;
        off = (off + 255) & ~(size_t)255;
        return p;
    };
    float* dis = (float*)carve((size_t)N * sizeof(float));
    float* h1  = (float*)carve((size_t)N * HID_DIM * sizeof(float));
    float* g1  = (float*)carve((size_t)N * HID_DIM * sizeof(float));
    float* h2  = (float*)carve((size_t)N * OUT_DIM * sizeof(float));

    const int TB = 256;

    // 1) symmetric normalization factors
    deg_init_kernel<<<(N + TB - 1) / TB, TB, 0, stream>>>(dis, N);
    deg_accum_kernel<<<(E + TB - 1) / TB, TB, 0, stream>>>(dst, dis, E);
    deg_rsqrt_kernel<<<(N + TB - 1) / TB, TB, 0, stream>>>(dis, N);

    // 2) h1 = x @ W1 (WMMA f16)
    {
        long long waves = (long long)(N / 16) * (HID_DIM / 16);
        long long blocks = (waves * 32 + TB - 1) / TB;
        gemm_wmma_f16<<<(int)blocks, TB, 0, stream>>>(x, W1, h1, N, IN_DIM, HID_DIM);
    }

    // 3) aggregate layer 1
    hipMemsetAsync(g1, 0, (size_t)N * HID_DIM * sizeof(float), stream);
    {
        long long threads = (long long)E * 32;
        scatter_f128<<<(int)((threads + TB - 1) / TB), TB, 0, stream>>>(src, dst, dis, h1, g1, E);
    }
    post1_kernel<<<(N * HID_DIM + TB - 1) / TB, TB, 0, stream>>>(g1, h1, dis, b1, N * HID_DIM);

    // 4) h2 = relu1 @ W2 (WMMA f16)
    {
        long long waves = (long long)(N / 16) * (OUT_DIM / 16);
        long long blocks = (waves * 32 + TB - 1) / TB;
        gemm_wmma_f16<<<(int)blocks, TB, 0, stream>>>(g1, W2, h2, N, HID_DIM, OUT_DIM);
    }

    // 5) aggregate layer 2 directly into d_out (init with self-loop + bias)
    init_out_kernel<<<(N * OUT_DIM + TB - 1) / TB, TB, 0, stream>>>(out, h2, dis, b2, N * OUT_DIM);
    {
        long long threads = (long long)E * 16;
        scatter_f64<<<(int)((threads + TB - 1) / TB), TB, 0, stream>>>(src, dst, dis, h2, out, E);
    }

    // 6) in-place log_softmax
    {
        long long threads = (long long)N * 32;
        logsoftmax64_kernel<<<(int)((threads + TB - 1) / TB), TB, 0, stream>>>(out, N);
    }
}
